// MHAIcoDecoder_7138235646307
// MI455X (gfx1250) — matrix operand order repaired and checked by reading, not compiled, not run on hardware
//
#include <hip/hip_runtime.h>
#include <hip/hip_bf16.h>

typedef __bf16 bf16;
typedef __attribute__((ext_vector_type(16))) __bf16 v16bf;
typedef __attribute__((ext_vector_type(8)))  float  v8f;

#define C_DIM 128
#define HD_DIM 64
#define KNB 6
#define ROW_STRIDE 136  // bf16 elems; 272B = 17*16B -> 16B-aligned chunks, conflict-free frags

// ---------------- WMMA fragment loaders (CDNA5 16-bit layouts) ----------------
// A-matrix 16x32 (MxK): lane M = lane&15, K-group offset = (lane&16)?8:0,
// vgpr pair j: K = kbase + (j<4?0:16) + 2*(j&3) + {0,1}  (two contiguous 16B runs)
__device__ __forceinline__ v16bf load_frag_A(const bf16* base, int lane, int kchunk, int ld) {
  const int m  = lane & 15;
  const int kb = kchunk * 32 + ((lane & 16) ? 8 : 0);
  const bf16* row = base + m * ld;
  v16bf a;
#pragma unroll
  for (int j = 0; j < 8; ++j) {
    const int k = kb + ((j < 4) ? 0 : 16) + 2 * (j & 3);
    a[2 * j]     = row[k];
    a[2 * j + 1] = row[k + 1];
  }
  return a;
}

// B-matrix 32x16 (KxN), B[k][n] = W[n][k] (D = X @ W^T):
// lane N = lane&15, K-group offset = (lane&16)?8:0, elements j<8: K = kb + j, j>=8: K = kb + 16 + (j-8)
__device__ __forceinline__ v16bf load_frag_B(const bf16* W, int lane, int kchunk, int nbase, int ld) {
  const int n  = nbase + (lane & 15);
  const int kb = kchunk * 32 + ((lane & 16) ? 8 : 0);
  const bf16* row = W + (size_t)n * ld + kb;
  v16bf b;
#pragma unroll
  for (int j = 0; j < 16; ++j) b[j] = row[(j < 8) ? j : j + 8];
  return b;
}

#define WMMA_BF16(A, B, Cacc) \
  __builtin_amdgcn_wmma_f32_16x16x32_bf16(false, (A), false, (B), (short)0, (Cacc), false, false)

// ---------------- async global -> LDS copy (CDNA5, ASYNCcnt) ----------------
__device__ __forceinline__ void async_b128(unsigned lds_byte, unsigned long long gaddr) {
  asm volatile("global_load_async_to_lds_b128 %0, %1, off" :: "v"(lds_byte), "v"(gaddr) : "memory");
}
__device__ __forceinline__ void wait_async() {
  asm volatile("s_wait_asynccnt 0" ::: "memory");
}
__device__ __forceinline__ unsigned lds_off(const void* p) {
  return (unsigned)(unsigned long long)p;  // LDS aperture keeps the offset in addr[31:0]
}

// ---------------- fp32 -> bf16 weight conversion ----------------
__global__ void cvt_kernel(const float* __restrict__ src, bf16* __restrict__ dst, int n) {
  int i = blockIdx.x * blockDim.x + threadIdx.x;
  if (i < n) dst[i] = (bf16)src[i];
}

// ---------------- embed: (B*12, 3) @ (3,128)^T + b -> bf16 ----------------
__global__ void embed_kernel(const float* __restrict__ x, const float* __restrict__ emb_w,
                             const float* __restrict__ emb_b, bf16* __restrict__ out, int total) {
  int i = blockIdx.x * blockDim.x + threadIdx.x;  // over total*128
  if (i >= total * C_DIM) return;
  int v = i >> 7, c = i & 127;
  const float* xv = x + v * 3;
  out[i] = (bf16)(emb_b[c] + xv[0] * emb_w[c * 3 + 0] + xv[1] * emb_w[c * 3 + 1] +
                  xv[2] * emb_w[c * 3 + 2]);
}

// ---------------- attention pooling: coarse -> fine level ----------------
__global__ __launch_bounds__(32) void pool_kernel(
    const bf16* __restrict__ x_in, bf16* __restrict__ x_out,
    const int* __restrict__ idx, const bf16* __restrict__ W1,
    const float* __restrict__ b1, const float* __restrict__ Vw, const float* __restrict__ Vb,
    int Vin, int Vout, int nTiles) {
  __shared__ bf16  sXn[96][ROW_STRIDE];
  __shared__ float sH[96][68];
  __shared__ int   sIdx[96];
  __shared__ float sScore[96];
  __shared__ float sAttn[16][KNB];

  const int tid  = threadIdx.x;
  const int tile = blockIdx.x % nTiles;
  const int b    = blockIdx.x / nTiles;
  const int v0   = tile * 16;

  for (int r = tid; r < 96; r += 32) {
    int m = v0 + r / KNB; if (m >= Vout) m = Vout - 1;
    sIdx[r] = idx[m * KNB + (r % KNB)];
  }
  __syncthreads();

  // async gather: 96 rows x 256B, in 16B chunks (48 per lane)
  {
    const bf16* xb = x_in + (size_t)b * Vin * C_DIM;
    const unsigned lbase = lds_off(&sXn[0][0]);
    for (int e = tid; e < 96 * 16; e += 32) {
      const int r = e >> 4, ch = e & 15;
      const unsigned laddr = lbase + (unsigned)(r * (ROW_STRIDE * 2) + ch * 16);
      const unsigned long long ga =
          (unsigned long long)(xb + (size_t)sIdx[r] * C_DIM) + (unsigned)(ch * 16);
      async_b128(laddr, ga);
    }
    wait_async();
  }
  __syncthreads();

  // Preload B fragments of W1 (N=64 -> 4 tiles, K=128 -> 4 chunks)
  v16bf Bf[4][4];
#pragma unroll
  for (int kt = 0; kt < 4; ++kt)
#pragma unroll
    for (int nt = 0; nt < 4; ++nt)
      Bf[kt][nt] = load_frag_B(W1, tid, kt, nt * 16, C_DIM);

  const int n_lo = tid & 15;
  const int m_hi = (tid & 16) ? 8 : 0;
  for (int mt = 0; mt < 6; ++mt) {
    v8f acc[4] = {};
#pragma unroll
    for (int kt = 0; kt < 4; ++kt) {
      v16bf a = load_frag_A(&sXn[mt * 16][0], tid, kt, ROW_STRIDE);
#pragma unroll
      for (int nt = 0; nt < 4; ++nt) acc[nt] = WMMA_BF16(a, Bf[kt][nt], acc[nt]);
    }
#pragma unroll
    for (int nt = 0; nt < 4; ++nt) {
      const int col = nt * 16 + n_lo;
      const float bias = b1[col];
#pragma unroll
      for (int r = 0; r < 8; ++r) sH[mt * 16 + m_hi + r][col] = tanhf(acc[nt][r] + bias);
    }
  }
  __syncthreads();

  for (int s = tid; s < 96; s += 32) {
    float d = Vb[0];
#pragma unroll 8
    for (int n = 0; n < HD_DIM; ++n) d += sH[s][n] * Vw[n];
    sScore[s] = d;
  }
  __syncthreads();
  if (tid < 16) {
    float mx = -1e30f;
    for (int k = 0; k < KNB; ++k) mx = fmaxf(mx, sScore[tid * KNB + k]);
    float e[KNB], sum = 0.f;
    for (int k = 0; k < KNB; ++k) { e[k] = __expf(sScore[tid * KNB + k] - mx); sum += e[k]; }
    const float inv = 1.f / sum;
    for (int k = 0; k < KNB; ++k) sAttn[tid][k] = e[k] * inv;
  }
  __syncthreads();
  bf16* xo = x_out + (size_t)b * Vout * C_DIM;
  for (int e = tid; e < 16 * C_DIM; e += 32) {
    int lm = e >> 7, c = e & 127;
    if (v0 + lm < Vout) {
      float a = 0.f;
      for (int k = 0; k < KNB; ++k) a += sAttn[lm][k] * (float)sXn[lm * KNB + k][c];
      xo[(size_t)(v0 + lm) * C_DIM + c] = (bf16)a;
    }
  }
}

// ---------------- per-level MHA (dh = 1 -> per-channel softmax over K=6) ----------------
__global__ __launch_bounds__(32) void mha_kernel(
    const bf16* __restrict__ x_in, bf16* __restrict__ x_out,
    const int* __restrict__ idx,
    const bf16* __restrict__ Wqkv, const float* __restrict__ bqkv,
    const bf16* __restrict__ Wout, const float* __restrict__ bout,
    int V, int nTiles) {
  __shared__ bf16 sXq[16][ROW_STRIDE];
  __shared__ bf16 sXn[96][ROW_STRIDE];
  __shared__ bf16 sQ[16][C_DIM];
  __shared__ bf16 sK[96][C_DIM];
  __shared__ bf16 sVv[96][C_DIM];
  __shared__ bf16 sO[16][ROW_STRIDE];
  __shared__ int  sIdx[96];

  const int tid  = threadIdx.x;
  const int tile = blockIdx.x % nTiles;
  const int b    = blockIdx.x / nTiles;
  const int v0   = tile * 16;
  const bf16* xb = x_in + (size_t)b * V * C_DIM;

  for (int r = tid; r < 96; r += 32) {
    int m = v0 + r / KNB; if (m >= V) m = V - 1;
    sIdx[r] = idx[m * KNB + (r % KNB)];
  }
  __syncthreads();

  // async copy: 16 query rows (8 chunks/lane) + 96 gathered rows (48 chunks/lane)
  {
    const unsigned lqbase = lds_off(&sXq[0][0]);
    for (int e = tid; e < 16 * 16; e += 32) {
      const int r = e >> 4, ch = e & 15;
      int m = v0 + r; if (m >= V) m = V - 1;
      async_b128(lqbase + (unsigned)(r * (ROW_STRIDE * 2) + ch * 16),
                 (unsigned long long)(xb + (size_t)m * C_DIM) + (unsigned)(ch * 16));
    }
    const unsigned lnbase = lds_off(&sXn[0][0]);
    for (int e = tid; e < 96 * 16; e += 32) {
      const int r = e >> 4, ch = e & 15;
      async_b128(lnbase + (unsigned)(r * (ROW_STRIDE * 2) + ch * 16),
                 (unsigned long long)(xb + (size_t)sIdx[r] * C_DIM) + (unsigned)(ch * 16));
    }
    wait_async();
  }
  __syncthreads();

  const int n_lo = tid & 15;
  const int m_hi = (tid & 16) ? 8 : 0;

  // Q = Xq @ Wq^T + bq  (16x128): hoist A fragments
  {
    v16bf aq[4];
#pragma unroll
    for (int kt = 0; kt < 4; ++kt) aq[kt] = load_frag_A(&sXq[0][0], tid, kt, ROW_STRIDE);
#pragma unroll
    for (int nt = 0; nt < 8; ++nt) {
      v8f acc = {};
#pragma unroll
      for (int kt = 0; kt < 4; ++kt)
        acc = WMMA_BF16(aq[kt], load_frag_B(Wqkv, tid, kt, nt * 16, C_DIM), acc);
      const int col = nt * 16 + n_lo;
      const float bias = bqkv[col];
#pragma unroll
      for (int r = 0; r < 8; ++r) sQ[m_hi + r][col] = (bf16)(acc[r] + bias);
    }
  }

  // K and V = Xn @ W{k,v}^T + b  (96x128 each): hoist A fragments per M-tile
  for (int kv = 0; kv < 2; ++kv) {
    const bf16*  W  = Wqkv + (size_t)(kv + 1) * C_DIM * C_DIM;
    const float* bp = bqkv + (kv + 1) * C_DIM;
    bf16 (*dst)[C_DIM] = (kv == 0) ? sK : sVv;
    for (int mt = 0; mt < 6; ++mt) {
      v16bf am[4];
#pragma unroll
      for (int kt = 0; kt < 4; ++kt) am[kt] = load_frag_A(&sXn[mt * 16][0], tid, kt, ROW_STRIDE);
#pragma unroll
      for (int nt = 0; nt < 8; ++nt) {
        v8f acc = {};
#pragma unroll
        for (int kt = 0; kt < 4; ++kt)
          acc = WMMA_BF16(am[kt], load_frag_B(W, tid, kt, nt * 16, C_DIM), acc);
        const int col = nt * 16 + n_lo;
        const float bias = bp[col];
#pragma unroll
        for (int r = 0; r < 8; ++r) dst[mt * 16 + m_hi + r][col] = (bf16)(acc[r] + bias);
      }
    }
  }
  __syncthreads();

  // per-(vertex, channel) softmax over K=6, scale = 1/sqrt(dh=1) = 1
  for (int e = tid; e < 16 * C_DIM; e += 32) {
    int lm = e >> 7, c = e & 127;
    const float q = (float)sQ[lm][c];
    float s[KNB], mx = -1e30f;
    for (int k = 0; k < KNB; ++k) { s[k] = q * (float)sK[lm * KNB + k][c]; mx = fmaxf(mx, s[k]); }
    float sum = 0.f;
    for (int k = 0; k < KNB; ++k) { s[k] = __expf(s[k] - mx); sum += s[k]; }
    const float inv = 1.f / sum;
    float o = 0.f;
    for (int k = 0; k < KNB; ++k) o += s[k] * inv * (float)sVv[lm * KNB + k][c];
    sO[lm][c] = (bf16)o;
  }
  __syncthreads();

  // x_out = O @ Wout^T + bout
  {
    v16bf ao[4];
#pragma unroll
    for (int kt = 0; kt < 4; ++kt) ao[kt] = load_frag_A(&sO[0][0], tid, kt, ROW_STRIDE);
    bf16* xo = x_out + (size_t)b * V * C_DIM;
#pragma unroll
    for (int nt = 0; nt < 8; ++nt) {
      v8f acc = {};
#pragma unroll
      for (int kt = 0; kt < 4; ++kt)
        acc = WMMA_BF16(ao[kt], load_frag_B(Wout, tid, kt, nt * 16, C_DIM), acc);
      const int col = nt * 16 + n_lo;
      const float bias = bout[col];
#pragma unroll
      for (int r = 0; r < 8; ++r) {
        const int m = v0 + m_hi + r;
        if (m < V) xo[(size_t)m * C_DIM + col] = (bf16)(acc[r] + bias);
      }
    }
  }
}

// ---------------- head: (B*V,128) @ (128,3)^T + b ----------------
__global__ void head_kernel(const bf16* __restrict__ x, const float* __restrict__ out_w,
                            const float* __restrict__ out_b, float* __restrict__ out, int total) {
  int i = blockIdx.x * blockDim.x + threadIdx.x;  // over total*3
  if (i >= total * 3) return;
  int v = i / 3, o = i % 3;
  const bf16* xv = x + (size_t)v * C_DIM;
  const float* w = out_w + o * C_DIM;
  float acc = out_b[o];
#pragma unroll 8
  for (int c = 0; c < C_DIM; ++c) acc += (float)xv[c] * w[c];
  out[i] = acc;
}

extern "C" void kernel_launch(void* const* d_in, const int* in_sizes, int n_in,
                              void* d_out, int out_size, void* d_ws, size_t ws_size,
                              hipStream_t stream) {
  (void)in_sizes; (void)n_in; (void)out_size; (void)ws_size;
  static const int V_LVL[7] = {12, 42, 162, 642, 2562, 10242, 40962};
  const int B = 4, NL = 6;

  const float* x = (const float*)d_in[0];
  const int* idx_pool[6]; const int* idx_mha[6];
  for (int i = 0; i < NL; ++i) {
    idx_pool[i] = (const int*)d_in[1 + 2 * i];
    idx_mha[i]  = (const int*)d_in[2 + 2 * i];
  }
  const float* emb_w     = (const float*)d_in[13];
  const float* emb_b     = (const float*)d_in[14];
  const float* pool_W1   = (const float*)d_in[15];
  const float* pool_W1_b = (const float*)d_in[16];
  const float* pool_V    = (const float*)d_in[17];
  const float* pool_V_b  = (const float*)d_in[18];
  const float* mha_in_w  = (const float*)d_in[19];
  const float* mha_in_b  = (const float*)d_in[20];
  const float* mha_out_w = (const float*)d_in[21];
  const float* mha_out_b = (const float*)d_in[22];
  const float* out_w     = (const float*)d_in[23];
  const float* out_b     = (const float*)d_in[24];

  // workspace: two ping-pong bf16 activation buffers + bf16 weights
  char* ws = (char*)d_ws;
  const size_t maxAct = (size_t)B * 40962 * C_DIM;
  bf16* buf0 = (bf16*)ws; ws += maxAct * sizeof(bf16);
  bf16* buf1 = (bf16*)ws; ws += maxAct * sizeof(bf16);
  bf16* W1bf   = (bf16*)ws; ws += (size_t)NL * HD_DIM * C_DIM * sizeof(bf16);
  bf16* Wqkvbf = (bf16*)ws; ws += (size_t)NL * 3 * C_DIM * C_DIM * sizeof(bf16);
  bf16* Woutbf = (bf16*)ws; ws += (size_t)NL * C_DIM * C_DIM * sizeof(bf16);

  const int n1 = NL * HD_DIM * C_DIM, n2 = NL * 3 * C_DIM * C_DIM, n3 = NL * C_DIM * C_DIM;
  cvt_kernel<<<(n1 + 255) / 256, 256, 0, stream>>>(pool_W1, W1bf, n1);
  cvt_kernel<<<(n2 + 255) / 256, 256, 0, stream>>>(mha_in_w, Wqkvbf, n2);
  cvt_kernel<<<(n3 + 255) / 256, 256, 0, stream>>>(mha_out_w, Woutbf, n3);

  const int tot0 = B * V_LVL[0];
  embed_kernel<<<(tot0 * C_DIM + 255) / 256, 256, 0, stream>>>(x, emb_w, emb_b, buf0, tot0);

  for (int i = 0; i < NL; ++i) {
    const int Vin = V_LVL[i], Vout = V_LVL[i + 1];
    const int nT = (Vout + 15) / 16;
    pool_kernel<<<B * nT, 32, 0, stream>>>(
        buf0, buf1, idx_pool[i],
        W1bf + (size_t)i * HD_DIM * C_DIM, pool_W1_b + (size_t)i * HD_DIM,
        pool_V + (size_t)i * HD_DIM, pool_V_b + i,
        Vin, Vout, nT);
    mha_kernel<<<B * nT, 32, 0, stream>>>(
        buf1, buf0, idx_mha[i],
        Wqkvbf + (size_t)i * 3 * C_DIM * C_DIM, mha_in_b + (size_t)i * 3 * C_DIM,
        Woutbf + (size_t)i * C_DIM * C_DIM, mha_out_b + (size_t)i * C_DIM,
        Vout, nT);
  }

  const int totF = B * V_LVL[NL];
  head_kernel<<<(totF * 3 + 255) / 256, 256, 0, stream>>>(buf0, out_w, out_b, (float*)d_out, totF);
}
